// CorrOptDiMPUnique_85023172592331
// MI455X (gfx1250) — matrix operand order repaired and checked by reading, not compiled, not run on hardware
//
#include <hip/hip_runtime.h>
#include <hip/hip_bf16.h>

typedef __bf16 bf16_t;
typedef __attribute__((ext_vector_type(16))) __bf16 v16bf;
typedef __attribute__((ext_vector_type(8)))  __bf16 v8bf;
typedef __attribute__((ext_vector_type(8)))  float  v8f;
typedef __attribute__((ext_vector_type(4)))  unsigned int u32x4;
typedef __attribute__((ext_vector_type(8)))  int i32x8;
typedef __attribute__((ext_vector_type(4)))  int i32x4;

// Problem constants (fixed by the reference)
constexpr int S = 8, C = 256, H = 23, W = 23;
constexpr int NF  = H * W;        // 529
constexpr int KF  = C * 9;        // 2304  (multiple of 32)
constexpr int NFP = 544;          // 529 padded to multiple of 32
constexpr int NUM_ITER = 2;
constexpr int NUM_BINS = 10;
constexpr float BIN_DISP = 0.5f;
constexpr float BG_MASK_VAL = 0.0f;
constexpr float MIN_FILTER_REG = 1e-5f;
constexpr float STEPLEN_REG = 0.0f;

// GEMM tiling
constexpr int MACRO_M = 64;    // per-workgroup M
constexpr int MACRO_N = 128;   // per-workgroup N
constexpr int KSTEP   = 32;    // bf16 WMMA K
// TDM writes rows of 32 bf16 (16 DWORDs) + 4 DWORD pad -> 20 DWORD pitch.
// Bank stride 20 (mod 64) gives conflict-free ds_load_b128 fragment reads.
constexpr int APITCH  = 40;    // bf16 elements per LDS row (32 data + 8 pad)

// ---------------------------------------------------------------------------
// im2col of feat into two bf16 layouts:
//   Xcm[s][p][k]  (row = spatial pos p, K-contiguous)       lda = KF
//   Xrm[s][k][pp] (row = channel-tap k, p-contiguous, pad)  lda = NFP (zero pad)
// ---------------------------------------------------------------------------
__global__ void dimp_im2col(const float* __restrict__ feat,
                            bf16_t* __restrict__ Xcm,
                            bf16_t* __restrict__ Xrm) {
  size_t idx = (size_t)blockIdx.x * blockDim.x + threadIdx.x;
  size_t total = (size_t)S * KF * NFP;
  if (idx >= total) return;
  int pp = (int)(idx % NFP);
  size_t t = idx / NFP;
  int k = (int)(t % KF);
  int s = (int)(t / KF);
  float val = 0.0f;
  if (pp < NF) {
    int c = k / 9, j = k % 9, dy = j / 3, dx = j % 3;
    int y = pp / W, x = pp % W;
    int iy = y + dy - 1, ix = x + dx - 1;
    if (iy >= 0 && iy < H && ix >= 0 && ix < W)
      val = feat[(((size_t)s * C + c) * H + iy) * W + ix];
  }
  bf16_t b = (bf16_t)val;
  Xrm[((size_t)s * KF + k) * NFP + pp] = b;
  if (pp < NF) Xcm[((size_t)s * NF + pp) * KF + k] = b;
}

// fp32 -> bf16 copy (filter to WMMA A-operand)
__global__ void dimp_f2bf(const float* __restrict__ src,
                          bf16_t* __restrict__ dst, size_t n) {
  size_t idx = (size_t)blockIdx.x * blockDim.x + threadIdx.x;
  if (idx < n) dst[idx] = (bf16_t)src[idx];
}

// ---------------------------------------------------------------------------
// TDM 2D tile load: rows of 32 bf16 (contiguous K) x tile_rows, row stride
// ld_elems, into LDS at lds_off with 4-DWORD row padding. rows_rem clamps the
// tensor extent so edge tiles read zeros (ISA: OOB reads return 0).
// D# layout per CDNA5 ISA.
// This toolchain exposes the 6-arg builtin:
//   (uint32x4 g0, int32x8 g1, int32x4 g2, int32x4 g3, int32x8 extra, i32 cpol)
// ---------------------------------------------------------------------------
__device__ __forceinline__ void dimp_tdm_load_2d(unsigned lds_off,
                                                 const bf16_t* gaddr,
                                                 unsigned rows_rem,
                                                 unsigned tile_rows,
                                                 unsigned ld_elems) {
  unsigned long long ga = (unsigned long long)(uintptr_t)gaddr;
  u32x4 g0;
  g0[0] = 1u;                                            // count=1, user mode
  g0[1] = lds_off;                                       // lds_addr (bytes)
  g0[2] = (unsigned)(ga & 0xFFFFFFFFu);                  // global_addr[31:0]
  g0[3] = (unsigned)((ga >> 32) & 0x01FFFFFFu) | (2u << 30);  // addr hi | type=2
  i32x8 g1;
  // data_size=2B (code 1), pad_enable, pad_interval=16DW (code 3),
  // pad_amount=4DW (code 3)
  g1[0] = (int)((1u << 16) | (1u << 20) | (3u << 22) | (3u << 25));
  g1[1] = (int)((KSTEP & 0xFFFFu) << 16);                // tensor_dim0 lo16
  g1[2] = (int)((rows_rem & 0xFFFFu) << 16);             // dim0 hi | dim1 lo
  g1[3] = (int)((rows_rem >> 16) | ((unsigned)KSTEP << 16)); // dim1 hi | tile_dim0
  g1[4] = (int)(tile_rows & 0xFFFFu);                    // tile_dim1 | tile_dim2=0
  g1[5] = (int)ld_elems;                                 // tensor_dim0_stride lo32
  g1[6] = 0;                                             // stride0 hi | stride1 lo
  g1[7] = 0;                                             // stride1 hi
  i32x4 gz = {0, 0, 0, 0};
  i32x8 gz8 = {0, 0, 0, 0, 0, 0, 0, 0};
  __builtin_amdgcn_tensor_load_to_lds(g0, g1, gz, gz, gz8, 0);
}

// ---------------------------------------------------------------------------
// Batched NT GEMM, bf16 in / fp32 out:  C[m][n] = sum_k A[m][k] * B[n][k]
// Workgroup: 8 waves -> 64x128 macro-tile; wave (wm,wn) in 2x4 grid owns a
// 32x32 register tile = 4 f32 accumulators = 4 WMMAs per 32-K step.
// A/B macro-tiles are staged into LDS by the Tensor Data Mover (wave 0),
// double-buffered, synced with s_wait_tensorcnt + workgroup barrier.
// Fragment layouts per ISA 7.12.2:
//  A 16-bit 16x32: lane l -> row (l&15), kbase=(l>>4)*8, elems {kb..kb+7, kb+16..kb+23}
//  B 16-bit 32x16: lane l -> col (l&15), kbase=(l>>4)*8, elems {kb..kb+7, kb+16..kb+23}
//  C/D 32-bit 16x16: lane l -> col (l&15), VGPR v -> row v + (l>>4)*8
// Kp must be a multiple of 32 (buffers zero-padded to guarantee this).
// ---------------------------------------------------------------------------
__global__ __launch_bounds__(256)
void dimp_wmma_gemm_tdm(const bf16_t* __restrict__ A, int lda, size_t strideA,
                        const bf16_t* __restrict__ B, int ldb, size_t strideB,
                        float* __restrict__ Cm, int ldc, size_t strideC,
                        int M, int N, int Kp) {
  __shared__ __align__(16) bf16_t ldsA[2][MACRO_M * APITCH];  // 2 x 5 KB
  __shared__ __align__(16) bf16_t ldsB[2][MACRO_N * APITCH];  // 2 x 10 KB

  const int s = blockIdx.y;
  A += strideA * s; B += strideB * s; Cm += strideC * s;

  const int Nt = (N + MACRO_N - 1) / MACRO_N;
  const int tm = blockIdx.x / Nt, tn = blockIdx.x % Nt;
  const int m0 = tm * MACRO_M, n0 = tn * MACRO_N;

  const int waveid = threadIdx.x >> 5;
  const int lane = threadIdx.x & 31;
  const int wm = waveid >> 2;        // 0..1
  const int wn = waveid & 3;         // 0..3
  const int half = lane >> 4;        // 0 | 1
  const int l15  = lane & 15;

  const bf16_t* Ag = A + (size_t)m0 * lda;
  const bf16_t* Bg = B + (size_t)n0 * ldb;
  const unsigned rowsA = (unsigned)(M - m0);
  const unsigned rowsB = (unsigned)(N - n0);

  // prologue: fill buffer 0
  if (waveid == 0) {
    dimp_tdm_load_2d((unsigned)(uintptr_t)&ldsA[0][0], Ag, rowsA, MACRO_M,
                     (unsigned)lda);
    dimp_tdm_load_2d((unsigned)(uintptr_t)&ldsB[0][0], Bg, rowsB, MACRO_N,
                     (unsigned)ldb);
    __builtin_amdgcn_s_wait_tensorcnt(0);
  }
  __syncthreads();

  v8f c00 = {}; v8f c01 = {}; v8f c10 = {}; v8f c11 = {};

  int buf = 0;
  for (int ks = 0; ks < Kp; ks += KSTEP) {
    // prefetch next K-step into the other buffer (wave 0, async via TDM)
    if (waveid == 0 && ks + KSTEP < Kp) {
      dimp_tdm_load_2d((unsigned)(uintptr_t)&ldsA[buf ^ 1][0], Ag + ks + KSTEP,
                       rowsA, MACRO_M, (unsigned)lda);
      dimp_tdm_load_2d((unsigned)(uintptr_t)&ldsB[buf ^ 1][0], Bg + ks + KSTEP,
                       rowsB, MACRO_N, (unsigned)ldb);
    }

    // fragments from LDS (bank-conflict-free thanks to 20-DWORD pitch)
    const bf16_t* pa = &ldsA[buf][(wm * 32 + l15) * APITCH + (half << 3)];
    const bf16_t* pb = &ldsB[buf][(wn * 32 + l15) * APITCH + (half << 3)];
    v8bf a0l = *(const v8bf*)(pa);
    v8bf a0h = *(const v8bf*)(pa + 16);
    v8bf a1l = *(const v8bf*)(pa + 16 * APITCH);
    v8bf a1h = *(const v8bf*)(pa + 16 * APITCH + 16);
    v8bf b0l = *(const v8bf*)(pb);
    v8bf b0h = *(const v8bf*)(pb + 16);
    v8bf b1l = *(const v8bf*)(pb + 16 * APITCH);
    v8bf b1h = *(const v8bf*)(pb + 16 * APITCH + 16);
    v16bf a0, a1, b0, b1;
#pragma unroll
    for (int i = 0; i < 8; ++i) {
      a0[i] = a0l[i]; a0[i + 8] = a0h[i];
      a1[i] = a1l[i]; a1[i + 8] = a1h[i];
      b0[i] = b0l[i]; b0[i + 8] = b0h[i];
      b1[i] = b1l[i]; b1[i + 8] = b1h[i];
    }
    c00 = __builtin_amdgcn_wmma_f32_16x16x32_bf16(false, a0, false, b0,
                                                  (short)0, c00, false, false);
    c01 = __builtin_amdgcn_wmma_f32_16x16x32_bf16(false, a0, false, b1,
                                                  (short)0, c01, false, false);
    c10 = __builtin_amdgcn_wmma_f32_16x16x32_bf16(false, a1, false, b0,
                                                  (short)0, c10, false, false);
    c11 = __builtin_amdgcn_wmma_f32_16x16x32_bf16(false, a1, false, b1,
                                                  (short)0, c11, false, false);

    // next buffer ready (wave0 tensorcnt) AND everyone done with this buffer
    if (waveid == 0) __builtin_amdgcn_s_wait_tensorcnt(0);
    __syncthreads();
    buf ^= 1;
  }

  const int rbase = m0 + wm * 32 + (half << 3);
  const int cbase = n0 + wn * 32 + l15;
#pragma unroll
  for (int v = 0; v < 8; ++v) {
    int r0 = rbase + v, r1 = r0 + 16;
    int cA = cbase, cB = cbase + 16;
    if (r0 < M) {
      if (cA < N) Cm[(size_t)r0 * ldc + cA] = c00[v];
      if (cB < N) Cm[(size_t)r0 * ldc + cB] = c01[v];
    }
    if (r1 < M) {
      if (cA < N) Cm[(size_t)r1 * ldc + cA] = c10[v];
      if (cB < N) Cm[(size_t)r1 * ldc + cB] = c11[v];
    }
  }
}

// ---------------------------------------------------------------------------
// Analytic label/spatial maps from the 10-bin hat basis: after unfolding, the
// map value at (cy,cx,y,x) depends only on r = sqrt((y-cy)^2+(x-cx)^2)/BIN_DISP.
// ---------------------------------------------------------------------------
__device__ __forceinline__ void dimp_bins(float dy, float dx,
                                          const float* __restrict__ wl,
                                          const float* __restrict__ wsp,
                                          float& lab, float& sp) {
  float r = sqrtf(dy * dy + dx * dx) / BIN_DISP;
  float L = 0.f, P = 0.f;
#pragma unroll
  for (int b = 0; b < NUM_BINS - 1; ++b) {
    float h = fmaxf(0.f, 1.f - fabsf(r - (float)b));
    L += wl[b] * h; P += wsp[b] * h;
  }
  float h9 = fminf(fmaxf(1.f + (r - (float)(NUM_BINS - 1)), 0.f), 1.f);
  L += wl[NUM_BINS - 1] * h9; P += wsp[NUM_BINS - 1] * h9;
  lab = L; sp = P;
}

// scores -> mapped residuals (bf16, K-padded) + comb = spatial*mask (fp32)
__global__ void dimp_residual(const float* __restrict__ scores,
                              const float* __restrict__ wl,
                              const float* __restrict__ wsp,
                              bf16_t* __restrict__ Rbf,
                              float* __restrict__ comb) {
  size_t idx = (size_t)blockIdx.x * blockDim.x + threadIdx.x;
  size_t total = (size_t)S * NF * NFP;
  if (idx >= total) return;
  int pp = (int)(idx % NFP);
  size_t t = idx / NFP;
  int m = (int)(t % NF);
  int s = (int)(t / NF);
  size_t row = (size_t)s * NF + m;
  if (pp >= NF) { Rbf[row * NFP + pp] = (bf16_t)0.f; return; }
  int p = pp;
  int cy = m / W, cx = m % W, y = p / W, x = p % W;
  float lab, sp;
  dimp_bins((float)(y - cy), (float)(x - cx), wl, wsp, lab, sp);
  float sc = scores[row * NF + p];
  float tmv = ((m == p) ? (1.0f - BG_MASK_VAL) : 0.0f) + BG_MASK_VAL;
  float act = (1.f - tmv) * 0.5f * fabsf(sc) + (1.f + tmv) * 0.5f * sc;
  float sgn = (sc > 0.f) ? 1.f : ((sc < 0.f) ? -1.f : 0.f);
  float msk = (1.f - tmv) * 0.5f * sgn + (1.f + tmv) * 0.5f;
  float mr = msk * (sp * (sp * (act - lab)));
  Rbf[row * NFP + pp] = (bf16_t)mr;
  comb[row * NF + p] = sp * msk;
}

// filter_grad = G + reg*f (in place) and bf16 copy for the next GEMM
__global__ void dimp_grad_combine(float* __restrict__ G,
                                  const float* __restrict__ fin,
                                  bf16_t* __restrict__ fgbf,
                                  const float* __restrict__ freg) {
  size_t idx = (size_t)blockIdx.x * blockDim.x + threadIdx.x;
  size_t total = (size_t)S * NF * KF;
  if (idx >= total) return;
  float fr = freg[0];
  float reg = fmaxf(fr * fr, MIN_FILTER_REG * MIN_FILTER_REG);
  float fg = G[idx] + reg * fin[idx];
  G[idx] = fg;
  fgbf[idx] = (bf16_t)fg;
}

// per-row sum of squares: out[row] = sum_i X[row*rowlen+i]^2
__global__ void dimp_rowsq(const float* __restrict__ X,
                           float* __restrict__ out, int rowlen) {
  __shared__ float sm[256];
  size_t row = blockIdx.x;
  float acc = 0.f;
  for (int i = threadIdx.x; i < rowlen; i += blockDim.x) {
    float v = X[row * (size_t)rowlen + i];
    acc += v * v;
  }
  sm[threadIdx.x] = acc;
  __syncthreads();
  for (int o = 128; o > 0; o >>= 1) {
    if ((int)threadIdx.x < o) sm[threadIdx.x] += sm[threadIdx.x + o];
    __syncthreads();
  }
  if (threadIdx.x == 0) out[row] = sm[0];
}

// per-row sum of (comb*X)^2
__global__ void dimp_rowsq_mul(const float* __restrict__ X,
                               const float* __restrict__ Mul,
                               float* __restrict__ out, int rowlen) {
  __shared__ float sm[256];
  size_t row = blockIdx.x;
  float acc = 0.f;
  for (int i = threadIdx.x; i < rowlen; i += blockDim.x) {
    size_t o = row * (size_t)rowlen + i;
    float v = Mul[o] * X[o];
    acc += v * v;
  }
  sm[threadIdx.x] = acc;
  __syncthreads();
  for (int o = 128; o > 0; o >>= 1) {
    if ((int)threadIdx.x < o) sm[threadIdx.x] += sm[threadIdx.x + o];
    __syncthreads();
  }
  if (threadIdx.x == 0) out[row] = sm[0];
}

// f_out = f_in - exp(lsl)*alpha*filter_grad ; optional bf16 mirror for next iter
__global__ void dimp_update(const float* __restrict__ fin,
                            const float* __restrict__ G,
                            const float* __restrict__ num,
                            const float* __restrict__ den,
                            const float* __restrict__ freg,
                            const float* __restrict__ lsl,
                            float* __restrict__ fout,
                            bf16_t* __restrict__ wbf, int write_bf) {
  size_t idx = (size_t)blockIdx.x * blockDim.x + threadIdx.x;
  size_t total = (size_t)S * NF * KF;
  if (idx >= total) return;
  size_t row = idx / KF;
  float fr = freg[0];
  float reg = fmaxf(fr * fr, MIN_FILTER_REG * MIN_FILTER_REG);
  float an = num[row];
  float ad = fmaxf(den[row] + reg * an, 1e-8f) + STEPLEN_REG * an;
  float alpha = an / ad;
  float step = __expf(lsl[0]);
  float fo = fin[idx] - step * alpha * G[idx];
  fout[idx] = fo;
  if (write_bf) wbf[idx] = (bf16_t)fo;
}

// ---------------------------------------------------------------------------
extern "C" void kernel_launch(void* const* d_in, const int* in_sizes, int n_in,
                              void* d_out, int out_size, void* d_ws, size_t ws_size,
                              hipStream_t stream) {
  (void)in_sizes; (void)n_in; (void)out_size; (void)ws_size;
  const float* d_filt = (const float*)d_in[0];
  const float* d_feat = (const float*)d_in[1];
  const float* d_wl   = (const float*)d_in[2];
  const float* d_wsp  = (const float*)d_in[3];
  const float* d_lsl  = (const float*)d_in[4];
  const float* d_freg = (const float*)d_in[5];
  float* out = (float*)d_out;

  // carve scratch (256B aligned slots)
  char* ws = (char*)d_ws;
  size_t off = 0;
  auto take = [&](size_t bytes) -> void* {
    void* p = ws + off;
    off = (off + bytes + 255) & ~(size_t)255;
    return p;
  };
  bf16_t* Xcm   = (bf16_t*)take((size_t)S * NF * KF * 2);
  bf16_t* Xrm   = (bf16_t*)take((size_t)S * KF * NFP * 2);
  bf16_t* Wbf   = (bf16_t*)take((size_t)S * NF * KF * 2);
  float*  score = (float*) take((size_t)S * NF * NF * 4);
  bf16_t* Rbf   = (bf16_t*)take((size_t)S * NF * NFP * 2);
  float*  comb  = (float*) take((size_t)S * NF * NF * 4);
  float*  G     = (float*) take((size_t)S * NF * KF * 4);
  bf16_t* fgbf  = (bf16_t*)take((size_t)S * NF * KF * 2);
  float*  fcur  = (float*) take((size_t)S * NF * KF * 4);
  float*  anum  = (float*) take((size_t)S * NF * 4);
  float*  aden  = (float*) take((size_t)S * NF * 4);

  const size_t TOTF = (size_t)S * NF * KF;   // filter elements
  auto blocks = [](size_t n) { return (unsigned)((n + 255) / 256); };

  auto gemm = [&](const bf16_t* A, int lda, size_t sA,
                  const bf16_t* B, int ldb, size_t sB,
                  float* Cp, int ldc, size_t sC, int M, int N, int Kp) {
    int mt = (M + MACRO_M - 1) / MACRO_M;
    int nt = (N + MACRO_N - 1) / MACRO_N;
    dim3 grid((unsigned)(mt * nt), S);
    dimp_wmma_gemm_tdm<<<grid, 256, 0, stream>>>(A, lda, sA, B, ldb, sB,
                                                 Cp, ldc, sC, M, N, Kp);
  };

  // im2col (both layouts, zero-padded) + bf16 filter
  dimp_im2col<<<blocks((size_t)S * KF * NFP), 256, 0, stream>>>(d_feat, Xcm, Xrm);
  dimp_f2bf<<<blocks(TOTF), 256, 0, stream>>>(d_filt, Wbf, TOTF);

  for (int it = 0; it < NUM_ITER; ++it) {
    const float* fin = (it == 0) ? d_filt : fcur;
    float* fout = (it == NUM_ITER - 1) ? out : fcur;

    // scores = f @ X           (M=NF, N=NF, K=KF)
    gemm(Wbf, KF, (size_t)NF * KF, Xcm, KF, (size_t)NF * KF,
         score, NF, (size_t)NF * NF, NF, NF, KF);
    // residuals + comb
    dimp_residual<<<blocks((size_t)S * NF * NFP), 256, 0, stream>>>(
        score, d_wl, d_wsp, Rbf, comb);
    // filter_grad(raw) = R @ X^T  (M=NF, N=KF, K=NFP zero-padded)
    gemm(Rbf, NFP, (size_t)NF * NFP, Xrm, NFP, (size_t)KF * NFP,
         G, KF, (size_t)NF * KF, NF, KF, NFP);
    // += reg*f, bf16 mirror
    dimp_grad_combine<<<blocks(TOTF), 256, 0, stream>>>(G, fin, fgbf, d_freg);
    // alpha numerator
    dimp_rowsq<<<S * NF, 256, 0, stream>>>(G, anum, KF);
    // scores_grad(raw) = filter_grad @ X  (reuse score buffer)
    gemm(fgbf, KF, (size_t)NF * KF, Xcm, KF, (size_t)NF * KF,
         score, NF, (size_t)NF * NF, NF, NF, KF);
    // alpha denominator: sum (comb * scores_grad)^2
    dimp_rowsq_mul<<<S * NF, 256, 0, stream>>>(score, comb, aden, NF);
    // f update
    dimp_update<<<blocks(TOTF), 256, 0, stream>>>(
        fin, G, anum, aden, d_freg, d_lsl, fout, Wbf,
        (it < NUM_ITER - 1) ? 1 : 0);
  }
}
